// MusicgenAttention_22316650070434
// MI455X (gfx1250) — matrix operand order repaired and checked by reading, not compiled, not run on hardware
//
#include <hip/hip_runtime.h>
#include <hip/hip_bf16.h>

typedef __attribute__((ext_vector_type(16))) _Float16 v16h;
typedef __attribute__((ext_vector_type(8)))  _Float16 v8h;
typedef __attribute__((ext_vector_type(8)))  float    v8f;
typedef __attribute__((ext_vector_type(4)))  int      v4i;

union F16x16 { v16h v; v8h h[2]; };

constexpr int Bsz = 2, T = 2048, D = 1024, H = 16, HD = 64;
constexpr int Mtot = Bsz * T;  // 4096

// ---------------- CDNA5 feature probes (compile-safe guards) ----------------
#if defined(__has_builtin)
#if __has_builtin(__builtin_amdgcn_global_load_async_to_lds_b128)
#define HAVE_ASYNC_LDS 1
#endif
#endif
#ifndef HAVE_ASYNC_LDS
#define HAVE_ASYNC_LDS 0
#endif

#if HAVE_ASYNC_LDS
// probe-derived signature: (int4 addrspace(1)*, int4 addrspace(3)*, imm, imm)
typedef __attribute__((address_space(1))) v4i gv4i;
typedef __attribute__((address_space(3))) v4i lv4i;
__device__ inline void async_cp_b128(const void* g, void* lds) {
  // GLOBAL_LOAD_ASYNC_TO_LDS_B128 (tracked by ASYNCcnt)
  __builtin_amdgcn_global_load_async_to_lds_b128((gv4i*)g, (lv4i*)lds, 0, 0);
}
__device__ inline void wait_async0() {
#if defined(__has_builtin) && __has_builtin(__builtin_amdgcn_s_wait_asynccnt)
  __builtin_amdgcn_s_wait_asynccnt(0);
#else
  asm volatile("s_wait_asynccnt 0x0" ::: "memory");
#endif
}
#endif

__device__ inline v8f wmma_f16(v16h a, v16h b, v8f c) {
  // D = A(16x32 f16) * B(32x16 f16) + C(16x16 f32)
  return __builtin_amdgcn_wmma_f32_16x16x32_f16(false, a, false, b, (short)0, c,
                                                false, false);
}

__device__ inline v8h cvt8(const float* s) {
  v8h r;
#pragma unroll
  for (int i = 0; i < 8; ++i) r[i] = (_Float16)s[i];
  return r;
}

// ---------------------------------------------------------------------------
// WMMA GEMM:  C[m,n] = sum_k A[m,k] * W[n,k] + bias[n]
//   Block 128 thr (4 waves), tile 128(M) x 64(N), K-step 64.
//   Wave owns 32 M rows (2 M-frags) x full 64 N -> 16 WMMA per K-step.
//   A_HALF: A already f16 (async-staged when available) vs f32 (convert).
//   OUT_MODE 0: (acc+bias)*scale -> f16 head-split [B,H,T,HD]
//   OUT_MODE 1:  acc+bias        -> f32 row-major  [M,D]
// ---------------------------------------------------------------------------
template <bool A_HALF, int OUT_MODE>
__global__ __launch_bounds__(128) void gemm_wmma_kernel(
    const void* __restrict__ Aptr, const float* __restrict__ W,
    const float* __restrict__ bias, void* __restrict__ Cptr, float scale) {
  __shared__ alignas(32) _Float16 lA[128 * 64];  // 16 KB
  __shared__ alignas(32) _Float16 lB[64 * 64];   //  8 KB

  const int tid  = threadIdx.x;
  const int wave = tid >> 5;
  const int l    = tid & 31;
  const int hi   = l >> 4;
  const int ln   = l & 15;
  const int m0   = blockIdx.x * 128;
  const int n0   = blockIdx.y * 64;

  v8f acc[2][4];
#pragma unroll
  for (int mt = 0; mt < 2; ++mt)
#pragma unroll
    for (int i = 0; i < 4; ++i) acc[mt][i] = (v8f){};

  for (int k0 = 0; k0 < D; k0 += 64) {
    // ---- stage A: thread t handles row m0+t, 64 k values ----
    if (A_HALF) {
      const _Float16* src = (const _Float16*)Aptr + (size_t)(m0 + tid) * D + k0;
      _Float16* dst = &lA[tid * 64];
#if HAVE_ASYNC_LDS
#pragma unroll
      for (int i = 0; i < 8; ++i) async_cp_b128(src + i * 8, dst + i * 8);
#else
#pragma unroll
      for (int i = 0; i < 8; ++i) *(v8h*)(dst + i * 8) = *(const v8h*)(src + i * 8);
#endif
    } else {
      const float* src = (const float*)Aptr + (size_t)(m0 + tid) * D + k0;
      _Float16* dst = &lA[tid * 64];
#pragma unroll
      for (int i = 0; i < 8; ++i) *(v8h*)(dst + i * 8) = cvt8(src + i * 8);
    }
    // ---- stage B: W rows n0..n0+63; thread -> (row tid&63, half (tid>>6)*32)
    {
      const int br = tid & 63, bc = (tid >> 6) * 32;
      const float* src = W + (size_t)(n0 + br) * D + k0 + bc;
      _Float16* dst = &lB[br * 64 + bc];
#pragma unroll
      for (int i = 0; i < 4; ++i) *(v8h*)(dst + i * 8) = cvt8(src + i * 8);
    }
#if HAVE_ASYNC_LDS
    if (A_HALF) wait_async0();
#endif
    __syncthreads();

#pragma unroll
    for (int s = 0; s < 2; ++s) {  // two 32-wide K sub-steps
      F16x16 a[2];
#pragma unroll
      for (int mt = 0; mt < 2; ++mt) {
        const _Float16* ar = &lA[(32 * wave + 16 * mt + ln) * 64 + s * 32];
        a[mt].h[0] = *(const v8h*)(ar + hi * 8);
        a[mt].h[1] = *(const v8h*)(ar + 16 + hi * 8);
      }
#pragma unroll
      for (int i = 0; i < 4; ++i) {
        F16x16 bfr;  // B frag: N = 16*i+ln, K = hi*8 + e (e<8), 16 + hi*8 + (e-8)
        const _Float16* brp = &lB[(16 * i + ln) * 64 + s * 32];
        bfr.h[0] = *(const v8h*)(brp + hi * 8);
        bfr.h[1] = *(const v8h*)(brp + 16 + hi * 8);
#pragma unroll
        for (int mt = 0; mt < 2; ++mt)
          acc[mt][i] = wmma_f16(a[mt].v, bfr.v, acc[mt][i]);
      }
    }
    __syncthreads();
  }

  // ---- epilogue ----
#pragma unroll
  for (int mt = 0; mt < 2; ++mt) {
#pragma unroll
    for (int i = 0; i < 4; ++i) {
      const int n = n0 + 16 * i + ln;
      const float bn = bias[n];
#pragma unroll
      for (int r = 0; r < 8; ++r) {
        const int m = m0 + 32 * wave + 16 * mt + r + hi * 8;
        float val = acc[mt][i][r] + bn;
        if (OUT_MODE == 0) {
          val *= scale;
          const int b = m >> 11, t = m & (T - 1);
          const int hh = n >> 6, d = n & 63;
          ((_Float16*)Cptr)[((size_t)(b * H + hh) * T + t) * HD + d] = (_Float16)val;
        } else {
          ((float*)Cptr)[(size_t)m * D + n] = val;
        }
      }
    }
  }
}

// ---------------------------------------------------------------------------
// Flash attention: grid (T/64, H, B), 128 threads (4 waves).
// 64 query rows per block (16 per wave); keys streamed in 64-row chunks,
// double-buffered with async global->LDS copies for K.
// ---------------------------------------------------------------------------
__global__ __launch_bounds__(128) void attn_wmma_kernel(
    const _Float16* __restrict__ Qh, const _Float16* __restrict__ Kh,
    const _Float16* __restrict__ Vh, const float* __restrict__ mask,
    _Float16* __restrict__ Oh) {
  __shared__ alignas(32) _Float16 lK[2][64 * 64];   // 16 KB  K chunk row-major
  __shared__ alignas(32) _Float16 lVt[2][64 * 64];  // 16 KB  V chunk transposed
  __shared__ alignas(32) _Float16 lP[4][16 * 64];   //  8 KB  per-wave P repack

  const int tid  = threadIdx.x;
  const int wave = tid >> 5;
  const int l    = tid & 31;
  const int hi   = l >> 4;
  const int ln   = l & 15;
  const int b = blockIdx.z, h = blockIdx.y, q0 = blockIdx.x * 64;

  const size_t headoff = (size_t)(b * H + h) * T * HD;

  // Q A-fragments pinned in registers (2 k-chunks over HD=64)
  F16x16 aQ[2];
  {
    const _Float16* qrow = Qh + headoff + (size_t)(q0 + 16 * wave + ln) * HD;
#pragma unroll
    for (int c = 0; c < 2; ++c) {
      aQ[c].h[0] = *(const v8h*)(qrow + c * 32 + hi * 8);
      aQ[c].h[1] = *(const v8h*)(qrow + c * 32 + 16 + hi * 8);
    }
  }

  v8f o[4];
#pragma unroll
  for (int i = 0; i < 4; ++i) o[i] = (v8f){};
  float rmax[8], rsum[8];
#pragma unroll
  for (int r = 0; r < 8; ++r) { rmax[r] = -3.0e38f; rsum[r] = 0.0f; }

  const int srow = tid >> 1;        // 0..63 (key row in chunk)
  const int scol = (tid & 1) * 32;  // 0 / 32

  auto stage = [&](int buf, int j0) {
    // K: straight copy -> async path
    const _Float16* ksrc = Kh + headoff + (size_t)(j0 + srow) * HD + scol;
    _Float16* kdst = &lK[buf][srow * 64 + scol];
#if HAVE_ASYNC_LDS
#pragma unroll
    for (int i = 0; i < 4; ++i) async_cp_b128(ksrc + i * 8, kdst + i * 8);
#else
#pragma unroll
    for (int i = 0; i < 4; ++i) *(v8h*)(kdst + i * 8) = *(const v8h*)(ksrc + i * 8);
#endif
    // V: transposed scatter lVt[d*64 + j]
    const _Float16* vsrc = Vh + headoff + (size_t)(j0 + srow) * HD + scol;
    _Float16* vt = lVt[buf];
#pragma unroll
    for (int i = 0; i < 32; ++i) vt[(scol + i) * 64 + srow] = vsrc[i];
  };

  stage(0, 0);

  for (int c = 0; c < T / 64; ++c) {
    const int j0  = c * 64;
    const int buf = c & 1;
#if HAVE_ASYNC_LDS
    wait_async0();
#endif
    __syncthreads();
    if (c + 1 < T / 64) stage(buf ^ 1, j0 + 64);  // overlap next-chunk staging

    // ---- S = Q K^T (16 x 64) ----
    v8f s[4];
#pragma unroll
    for (int n = 0; n < 4; ++n) {
      v8f cc = (v8f){};
#pragma unroll
      for (int ds = 0; ds < 2; ++ds) {
        F16x16 bk;  // B frag: N = key 16*n+ln, K = d = ds*32 + table order (hi*8 / 16+hi*8)
        const _Float16* kr = &lK[buf][(16 * n + ln) * 64 + ds * 32];
        bk.h[0] = *(const v8h*)(kr + hi * 8);
        bk.h[1] = *(const v8h*)(kr + 16 + hi * 8);
        cc = wmma_f16(aQ[ds].v, bk.v, cc);
      }
      s[n] = cc;
    }

    // ---- add attention mask [B,1,T,T] + prefetch next chunk ----
#pragma unroll
    for (int n = 0; n < 4; ++n)
#pragma unroll
      for (int r = 0; r < 8; ++r) {
        const int q = q0 + 16 * wave + r + hi * 8;
        s[n][r] += mask[((size_t)b * T + q) * T + j0 + n * 16 + ln];
      }
    if (c + 1 < T / 64)
      __builtin_prefetch(&mask[((size_t)b * T + q0 + 16 * wave + ln) * T + j0 + 64], 0, 0);

    // ---- online softmax (reduce across 16-lane halves) ----
    _Float16* pw = lP[wave];
#pragma unroll
    for (int r = 0; r < 8; ++r) {
      float t0 = fmaxf(fmaxf(s[0][r], s[1][r]), fmaxf(s[2][r], s[3][r]));
#pragma unroll
      for (int off = 8; off > 0; off >>= 1) t0 = fmaxf(t0, __shfl_xor(t0, off, 32));
      const float mnew  = fmaxf(rmax[r], t0);
      const float alpha = __expf(rmax[r] - mnew);
      float p[4], ps = 0.0f;
#pragma unroll
      for (int n = 0; n < 4; ++n) { p[n] = __expf(s[n][r] - mnew); ps += p[n]; }
#pragma unroll
      for (int off = 8; off > 0; off >>= 1) ps += __shfl_xor(ps, off, 32);
      rsum[r] = rsum[r] * alpha + ps;
      rmax[r] = mnew;
#pragma unroll
      for (int i = 0; i < 4; ++i) o[i][r] *= alpha;
      // repack P: C layout (row r+8*hi, col 16*n+ln) -> row-major 16x64
#pragma unroll
      for (int n = 0; n < 4; ++n)
        pw[(r + hi * 8) * 64 + n * 16 + ln] = (_Float16)p[n];
    }
    asm volatile("s_wait_dscnt 0x0" ::: "memory");  // same-wave LDS RAW drain

    // ---- O += P @ V ----
    F16x16 aP[2];
    const _Float16* pr = pw + ln * 64;
#pragma unroll
    for (int cj = 0; cj < 2; ++cj) {
      aP[cj].h[0] = *(const v8h*)(pr + cj * 32 + hi * 8);
      aP[cj].h[1] = *(const v8h*)(pr + cj * 32 + 16 + hi * 8);
    }
#pragma unroll
    for (int i = 0; i < 4; ++i) {
#pragma unroll
      for (int cj = 0; cj < 2; ++cj) {
        F16x16 bv;  // B frag: N = d (16*i+ln), K = j = cj*32 + table order (hi*8 / 16+hi*8)
        const _Float16* vr = &lVt[buf][(16 * i + ln) * 64 + cj * 32];
        bv.h[0] = *(const v8h*)(vr + hi * 8);
        bv.h[1] = *(const v8h*)(vr + 16 + hi * 8);
        o[i] = wmma_f16(aP[cj].v, bv.v, o[i]);
      }
    }
  }

  // ---- normalize and store f16 in [B,T,D] (merged-head layout) ----
#pragma unroll
  for (int i = 0; i < 4; ++i) {
#pragma unroll
    for (int r = 0; r < 8; ++r) {
      const int q = q0 + 16 * wave + r + hi * 8;
      const float val = o[i][r] / rsum[r];
      Oh[((size_t)b * T + q) * D + h * HD + 16 * i + ln] = (_Float16)val;
    }
  }
}

// ---------------------------------------------------------------------------
extern "C" void kernel_launch(void* const* d_in, const int* in_sizes, int n_in,
                              void* d_out, int out_size, void* d_ws, size_t ws_size,
                              hipStream_t stream) {
  const float* hs   = (const float*)d_in[0];
  const float* mask = (const float*)d_in[1];
  const float* Wq   = (const float*)d_in[2];
  const float* bq   = (const float*)d_in[3];
  const float* Wk   = (const float*)d_in[4];
  const float* bk   = (const float*)d_in[5];
  const float* Wv   = (const float*)d_in[6];
  const float* bv   = (const float*)d_in[7];
  const float* Wo   = (const float*)d_in[8];
  const float* bo   = (const float*)d_in[9];

  const size_t elems = (size_t)Mtot * D;  // 4M f16 each
  _Float16* Qh = (_Float16*)d_ws;
  _Float16* Kh = Qh + elems;
  _Float16* Vh = Kh + elems;
  _Float16* Oh = Vh + elems;

  const dim3 gg(Mtot / 128, D / 64);  // 32 x 16 blocks

  // QKV projections (q pre-scaled by 1/sqrt(64))
  gemm_wmma_kernel<false, 0><<<gg, 128, 0, stream>>>(hs, Wq, bq, Qh, 0.125f);
  gemm_wmma_kernel<false, 0><<<gg, 128, 0, stream>>>(hs, Wk, bk, Kh, 1.0f);
  gemm_wmma_kernel<false, 0><<<gg, 128, 0, stream>>>(hs, Wv, bv, Vh, 1.0f);

  // streaming-softmax attention
  attn_wmma_kernel<<<dim3(T / 64, H, Bsz), 128, 0, stream>>>(Qh, Kh, Vh, mask, Oh);

  // output projection -> f32 d_out
  gemm_wmma_kernel<true, 1><<<gg, 128, 0, stream>>>(Oh, Wo, bo, d_out, 1.0f);
}
